// TwoViewGCL_39599598469172
// MI455X (gfx1250) — matrix operand order repaired and checked by reading, not compiled, not run on hardware
//
#include <hip/hip_runtime.h>
#include <hip/hip_bf16.h>

// ---------------- problem constants (match reference) ----------------
#define NNODES 100000
#define DLAT   256     // latent dim (K of first GEMM, N of first GEMM)
#define BATCH  4096    // batch nodes (M of all GEMMs)
#define KNB    32      // padded degree
#define DZ     128     // projection output dim
#define TEMP_INV (1.0f / 0.07f)

typedef __attribute__((ext_vector_type(16))) _Float16 v16h;
typedef __attribute__((ext_vector_type(8)))  float    v8f;

union Frag16 { v16h h; unsigned int u[8]; };

// ---------------- WMMA fragment loaders (wave32, ISA 7.12.2 layouts) --------
// A: 16x32 f16, row-major source A[m][k], lda in elements.
//   lanes 0-15 : M=lane,   K = {0..7, 16..23}
//   lanes 16-31: M=lane-16,K = {8..15,24..31}
__device__ __forceinline__ v16h load_a_frag(const _Float16* __restrict__ A,
                                            int lda, int m0, int k0) {
  const int lane = threadIdx.x & 31;
  const int half = lane >> 4;
  const _Float16* p = A + (size_t)(m0 + (lane & 15)) * lda + k0 + 8 * half;
  Frag16 f;
#pragma unroll
  for (int i = 0; i < 8; ++i) {
    const int k = (i < 4) ? (2 * i) : (16 + 2 * (i - 4)); // even -> dword aligned pair
    f.u[i] = *reinterpret_cast<const unsigned int*>(p + k);
  }
  return f.h;
}

// B: 32x16 f16 where B[k][n] = Bt[n][k] (Bt row-major, ldb in elements).
//   lanes 0-15 : N=lane,    K = {0..7, 16..23}
//   lanes 16-31: N=lane-16, K = {8..15,24..31}
__device__ __forceinline__ v16h load_b_frag(const _Float16* __restrict__ Bt,
                                            int ldb, int n0, int k0) {
  const int lane = threadIdx.x & 31;
  const int half = lane >> 4;
  const _Float16* p = Bt + (size_t)(n0 + (lane & 15)) * ldb + k0 + 8 * half;
  Frag16 f;
#pragma unroll
  for (int i = 0; i < 8; ++i)
    f.u[i] = *reinterpret_cast<const unsigned int*>(p + ((i < 4) ? (2 * i) : (16 + 2 * (i - 4))));
  return f.h;
}

// ---------------- async global->LDS helpers (CDNA5, ASYNCcnt path) ----------
// ptrtoint of an addrspace(3) pointer yields the wave-relative LDS byte
// address, which is what GLOBAL_LOAD_ASYNC_TO_LDS takes in its VDST VGPR.
__device__ __forceinline__ unsigned lds_addr_of(const void* p) {
  return (unsigned)(unsigned long long)(__attribute__((address_space(3))) const void*)p;
}

__device__ __forceinline__ void async_load_b128(unsigned lds_off, const void* gaddr) {
  asm volatile("global_load_async_to_lds_b128 %0, %1, off"
               :: "v"(lds_off), "v"(gaddr)
               : "memory");
}

__device__ __forceinline__ void wait_asynccnt0() {
  asm volatile("s_wait_asynccnt 0x0" ::: "memory");
}

// ---------------- kernel: zero accumulators ----------------
__global__ void zero_kernel(float* p, int n) {
  int i = blockIdx.x * blockDim.x + threadIdx.x;
  if (i < n) p[i] = 0.0f;
}

// ---------------- kernel: f32 weight -> transposed f16 ----------------
// W is [K x N] row-major f32; Wt is [N x K] row-major f16 (so B-frag loads
// read packed consecutive-K pairs).
__global__ void convert_w_kernel(const float* __restrict__ W,
                                 _Float16* __restrict__ Wt, int K, int N) {
  int idx = blockIdx.x * blockDim.x + threadIdx.x;
  if (idx < K * N) {
    int k = idx / N, n = idx - k * N;
    Wt[(size_t)n * K + k] = (_Float16)W[idx];
  }
}

// ---------------- kernel: masked neighbor-mean aggregation ----------------
// grid (BATCH, 2), block DLAT. Writes f16 [BATCH x DLAT] per view.
__global__ void aggregate_kernel(const float* __restrict__ nf,
                                 const float* __restrict__ rel,
                                 const int* __restrict__ nb1, const int* __restrict__ rl1,
                                 const unsigned char* __restrict__ mk1,
                                 const int* __restrict__ nb2, const int* __restrict__ rl2,
                                 const unsigned char* __restrict__ mk2,
                                 const int* __restrict__ self_ids,
                                 _Float16* __restrict__ out1,
                                 _Float16* __restrict__ out2) {
  const int b = blockIdx.x;
  const int view = blockIdx.y;
  const int d = threadIdx.x;
  const int* nb = (view == 0) ? nb1 : nb2;
  const int* rl = (view == 0) ? rl1 : rl2;
  const unsigned char* mk = (view == 0) ? mk1 : mk2;
  _Float16* out = (view == 0) ? out1 : out2;

  __shared__ int s_nb[KNB];
  __shared__ int s_rl[KNB];
  __shared__ int s_mk[KNB];
  if (d < KNB) {
    s_nb[d] = nb[b * KNB + d];
    s_rl[d] = rl[b * KNB + d];
    s_mk[d] = (int)mk[b * KNB + d];
  }
  __syncthreads();

  float acc = 0.0f;
  int cnt = 0;
#pragma unroll 4
  for (int k = 0; k < KNB; ++k) {
    if (s_mk[k]) {   // uniform across block (shared scalar)
      acc += nf[(size_t)s_nb[k] * DLAT + d] + rel[(size_t)s_rl[k] * DLAT + d];
      ++cnt;
    }
  }
  float v;
  if (cnt > 0) v = acc / (float)cnt;
  else         v = nf[(size_t)self_ids[b] * DLAT + d];
  out[(size_t)b * DLAT + d] = (_Float16)v;
}

// ---------------- kernel: WMMA GEMM  out = act(A @ Bt^T + bias) ------------
// A: [M x K] f16 row-major. Bt: [N x K] f16 row-major (i.e. B transposed).
// 8 waves per block tiled 2 (M) x 4 (N) -> 32x64 output tile per block.
// RELU_F16_OUT: outH[m*N+n] = relu(v) as f16; else outF[m*N+n] = v as f32.
template <bool RELU_F16_OUT>
__global__ void gemm_kernel(const _Float16* __restrict__ A,
                            const _Float16* __restrict__ Bt,
                            const float* __restrict__ bias,
                            _Float16* __restrict__ outH,
                            float* __restrict__ outF,
                            int M, int N, int K) {
  const int lane = threadIdx.x & 31;
  const int w  = threadIdx.x >> 5;
  const int wy = w >> 2, wx = w & 3;
  const int m0 = blockIdx.x * 32 + wy * 16;
  const int n0 = blockIdx.y * 64 + wx * 16;

  v8f c = {};
  for (int k0 = 0; k0 < K; k0 += 32) {
    if (k0 + 32 < K) {
      __builtin_prefetch(A + (size_t)(m0 + (lane & 15)) * K + k0 + 32, 0, 0);
      __builtin_prefetch(Bt + (size_t)(n0 + (lane & 15)) * K + k0 + 32, 0, 0);
    }
    v16h a = load_a_frag(A, K, m0, k0);
    v16h b = load_b_frag(Bt, K, n0, k0);
    c = __builtin_amdgcn_wmma_f32_16x16x32_f16(false, a, false, b,
                                               (short)0, c, false, false);
  }

  const int half = lane >> 4;
  const int n = n0 + (lane & 15);
  const float bv = bias[n];
#pragma unroll
  for (int r = 0; r < 8; ++r) {
    const int m = m0 + r + 8 * half;
    float v = c[r] + bv;
    if (RELU_F16_OUT) outH[(size_t)m * N + n] = (_Float16)fmaxf(v, 0.0f);
    else              outF[(size_t)m * N + n] = v;
  }
}

// ---------------- kernel: row L2-normalize, f32 -> f16 ----------------
// grid BATCH, block DZ (=128, 4 waves).
__global__ void normalize_kernel(const float* __restrict__ Z,
                                 _Float16* __restrict__ Zh) {
  const int row = blockIdx.x;
  const int t = threadIdx.x;
  float v = Z[(size_t)row * DZ + t];
  float ss = v * v;
#pragma unroll
  for (int o = 16; o > 0; o >>= 1) ss += __shfl_xor(ss, o, 32);
  __shared__ float ws[4];
  if ((t & 31) == 0) ws[t >> 5] = ss;
  __syncthreads();
  float tot = ws[0] + ws[1] + ws[2] + ws[3];
  Zh[(size_t)row * DZ + t] = (_Float16)(v * rsqrtf(fmaxf(tot, 1e-30f)));
}

// ---------------- kernel: fused sim GEMM + softmax statistics ----------------
// sim[i][j] = (zn1[i] . zn2[j]) / T.  Per block: 32x64 tile via 8 WMMAs x 4
// K-steps.  A panel (32x128) and B panel (64x128) are staged into LDS with
// CDNA5 async global->LDS B128 copies (ASYNCcnt), then WMMA operands are read
// from LDS.  Accumulates sum(exp(sim)) per row and per column (atomics) and
// captures the diagonal.
__global__ void sim_kernel(const _Float16* __restrict__ Z1,
                           const _Float16* __restrict__ Z2,
                           float* __restrict__ rowS,
                           float* __restrict__ colS,
                           float* __restrict__ diag) {
  __shared__ _Float16 As[32 * DZ];       // 8 KB: rows  gRow0..gRow0+31, K=0..127
  __shared__ _Float16 Bs[64 * DZ];       // 16 KB: rows gCol0..gCol0+63, K=0..127
  __shared__ float tile[2][4][16][16];   // raw sim values

  const int t = threadIdx.x;
  const int lane = t & 31;
  const int w  = t >> 5;
  const int wy = w >> 2, wx = w & 3;
  const int gRow0 = blockIdx.x * 32;
  const int gCol0 = blockIdx.y * 64;

  // ---- async stage: each row is 256B = 16 x B128 chunks ----
#pragma unroll
  for (int it = 0; it < 2; ++it) {       // A: 512 chunks over 256 threads
    const int idx = t + it * 256;
    const int row = idx >> 4, chunk = idx & 15;
    async_load_b128(lds_addr_of(&As[row * DZ + chunk * 8]),
                    Z1 + (size_t)(gRow0 + row) * DZ + chunk * 8);
  }
#pragma unroll
  for (int it = 0; it < 4; ++it) {       // B: 1024 chunks over 256 threads
    const int idx = t + it * 256;
    const int row = idx >> 4, chunk = idx & 15;
    async_load_b128(lds_addr_of(&Bs[row * DZ + chunk * 8]),
                    Z2 + (size_t)(gCol0 + row) * DZ + chunk * 8);
  }
  wait_asynccnt0();      // my transfers landed in LDS
  __syncthreads();       // everyone's transfers landed

  // ---- WMMA over LDS panels ----
  v8f c = {};
#pragma unroll
  for (int k0 = 0; k0 < DZ; k0 += 32) {
    v16h a = load_a_frag(As, DZ, wy * 16, k0);
    v16h b = load_b_frag(Bs, DZ, wx * 16, k0);
    c = __builtin_amdgcn_wmma_f32_16x16x32_f16(false, a, false, b,
                                               (short)0, c, false, false);
  }

  const int half = lane >> 4;
  const int nc = lane & 15;
#pragma unroll
  for (int r = 0; r < 8; ++r)
    tile[wy][wx][r + 8 * half][nc] = c[r] * TEMP_INV;
  __syncthreads();

  if (t < 32) {
    // one of the 32 tile rows: sum exp over 64 cols
    float s = 0.0f;
#pragma unroll
    for (int x = 0; x < 4; ++x)
#pragma unroll
      for (int n = 0; n < 16; ++n)
        s += __expf(tile[t >> 4][x][t & 15][n]);
    atomicAdd(&rowS[gRow0 + t], s);
    const int i = gRow0 + t;
    if (i >= gCol0 && i < gCol0 + 64) {
      const int cc = i - gCol0;
      diag[i] = tile[t >> 4][cc >> 4][t & 15][cc & 15];
    }
  } else if (t < 96) {
    // one of the 64 tile cols: sum exp over 32 rows
    const int cidx = t - 32;
    float s = 0.0f;
#pragma unroll
    for (int y = 0; y < 2; ++y)
#pragma unroll
      for (int m = 0; m < 16; ++m)
        s += __expf(tile[y][cidx >> 4][m][cidx & 15]);
    atomicAdd(&colS[gCol0 + cidx], s);
  }
}

// ---------------- kernel: final scalar reduce ----------------
// loss = mean_i( 0.5*(log(rowS_i) + log(colS_i)) - diag_i )
__global__ void finalize_kernel(const float* __restrict__ rowS,
                                const float* __restrict__ colS,
                                const float* __restrict__ diag,
                                float* __restrict__ out) {
  float acc = 0.0f;
  for (int i = threadIdx.x; i < BATCH; i += blockDim.x)
    acc += 0.5f * (logf(rowS[i]) + logf(colS[i])) - diag[i];
  __shared__ float sm[256];
  sm[threadIdx.x] = acc;
  __syncthreads();
  for (int s = 128; s > 0; s >>= 1) {
    if (threadIdx.x < s) sm[threadIdx.x] += sm[threadIdx.x + s];
    __syncthreads();
  }
  if (threadIdx.x == 0) out[0] = sm[0] / (float)BATCH;
}

// ---------------- host launcher ----------------
extern "C" void kernel_launch(void* const* d_in, const int* in_sizes, int n_in,
                              void* d_out, int out_size, void* d_ws, size_t ws_size,
                              hipStream_t stream) {
  const float* nf   = (const float*)d_in[0];   // [N, D]
  const float* rel  = (const float*)d_in[1];   // [9, D]
  const int*   nb1  = (const int*)d_in[2];
  const int*   rl1  = (const int*)d_in[3];
  const unsigned char* mk1 = (const unsigned char*)d_in[4];
  const int*   nb2  = (const int*)d_in[5];
  const int*   rl2  = (const int*)d_in[6];
  const unsigned char* mk2 = (const unsigned char*)d_in[7];
  const int*   self_ids = (const int*)d_in[8];
  const float* W1a = (const float*)d_in[9];
  const float* b1a = (const float*)d_in[10];
  const float* W1b = (const float*)d_in[11];
  const float* b1b = (const float*)d_in[12];
  const float* W2a = (const float*)d_in[13];
  const float* b2a = (const float*)d_in[14];
  const float* W2b = (const float*)d_in[15];
  const float* b2b = (const float*)d_in[16];

  // ---- workspace bump allocator (256B aligned) ----
  char* ws = (char*)d_ws;
  auto alloc = [&](size_t bytes) -> void* {
    void* p = (void*)ws;
    ws += (bytes + 255) & ~(size_t)255;
    return p;
  };
  _Float16* W1a_t = (_Float16*)alloc((size_t)DLAT * DLAT * 2);
  _Float16* W1b_t = (_Float16*)alloc((size_t)DLAT * DZ * 2);
  _Float16* W2a_t = (_Float16*)alloc((size_t)DLAT * DLAT * 2);
  _Float16* W2b_t = (_Float16*)alloc((size_t)DLAT * DZ * 2);
  // per-view ping-pong buffers (2 MB each): A holds v(f16) then z(f32);
  // B holds h(f16) then zn(f16)
  void* bufA1 = alloc((size_t)BATCH * DLAT * 2);
  void* bufB1 = alloc((size_t)BATCH * DLAT * 2);
  void* bufA2 = alloc((size_t)BATCH * DLAT * 2);
  void* bufB2 = alloc((size_t)BATCH * DLAT * 2);
  float* accum = (float*)alloc((size_t)3 * BATCH * 4);
  float* rowS = accum;
  float* colS = accum + BATCH;
  float* diag = accum + 2 * BATCH;

  _Float16* v1h  = (_Float16*)bufA1;
  _Float16* v2h  = (_Float16*)bufA2;
  _Float16* h1h  = (_Float16*)bufB1;
  _Float16* h2h  = (_Float16*)bufB2;
  float*    z1   = (float*)bufA1;     // overwrites v1h (no longer needed)
  float*    z2   = (float*)bufA2;
  _Float16* zn1h = (_Float16*)bufB1;  // overwrites h1h (no longer needed)
  _Float16* zn2h = (_Float16*)bufB2;

  // 1) zero softmax accumulators
  zero_kernel<<<(3 * BATCH + 255) / 256, 256, 0, stream>>>(accum, 3 * BATCH);

  // 2) weights -> transposed f16
  convert_w_kernel<<<(DLAT * DLAT + 255) / 256, 256, 0, stream>>>(W1a, W1a_t, DLAT, DLAT);
  convert_w_kernel<<<(DLAT * DZ   + 255) / 256, 256, 0, stream>>>(W1b, W1b_t, DLAT, DZ);
  convert_w_kernel<<<(DLAT * DLAT + 255) / 256, 256, 0, stream>>>(W2a, W2a_t, DLAT, DLAT);
  convert_w_kernel<<<(DLAT * DZ   + 255) / 256, 256, 0, stream>>>(W2b, W2b_t, DLAT, DZ);

  // 3) aggregation (both views)
  aggregate_kernel<<<dim3(BATCH, 2), DLAT, 0, stream>>>(
      nf, rel, nb1, rl1, mk1, nb2, rl2, mk2, self_ids, v1h, v2h);

  // 4) h = relu(v @ W*a + b*a)   [4096 x 256], K=256
  dim3 g1(BATCH / 32, DLAT / 64);
  gemm_kernel<true><<<g1, 256, 0, stream>>>(v1h, W1a_t, b1a, h1h, nullptr,
                                            BATCH, DLAT, DLAT);
  gemm_kernel<true><<<g1, 256, 0, stream>>>(v2h, W2a_t, b2a, h2h, nullptr,
                                            BATCH, DLAT, DLAT);

  // 5) z = h @ W*b + b*b   [4096 x 128], K=256
  dim3 g2(BATCH / 32, DZ / 64);
  gemm_kernel<false><<<g2, 256, 0, stream>>>(h1h, W1b_t, b1b, nullptr, z1,
                                             BATCH, DZ, DLAT);
  gemm_kernel<false><<<g2, 256, 0, stream>>>(h2h, W2b_t, b2b, nullptr, z2,
                                             BATCH, DZ, DLAT);

  // 6) row-normalize to unit vectors (f16 out)
  normalize_kernel<<<BATCH, DZ, 0, stream>>>(z1, zn1h);
  normalize_kernel<<<BATCH, DZ, 0, stream>>>(z2, zn2h);

  // 7) fused sim GEMM (LDS-staged via async copies) + softmax statistics
  sim_kernel<<<dim3(BATCH / 32, BATCH / 64), 256, 0, stream>>>(
      zn1h, zn2h, rowS, colS, diag);

  // 8) scalar loss
  finalize_kernel<<<1, 256, 0, stream>>>(rowS, colS, diag, (float*)d_out);
}
